// MTGN_76510547411028
// MI455X (gfx1250) — matrix operand order repaired and checked by reading, not compiled, not run on hardware
//
#include <hip/hip_runtime.h>
#include <math.h>
#include <stdint.h>

// ---------------------------------------------------------------- constants
constexpr int BG    = 64;                 // graphs
constexpr int NPG   = 300;                // nodes per graph
constexpr int NN    = BG * NPG;           // 19200 nodes
constexpr int EDG   = 153600;             // real edges (no self loops)
constexpr int DIN   = 1024;
constexpr int DD    = 512;
constexpr int NC    = 320;
constexpr int HH    = 4;                  // GAT heads
constexpr int DGH   = 128;                // per-head dim
constexpr int KPOOL = 240;                // top-k per graph
constexpr float ALPHA = 0.3f;
constexpr float BETA1 = 0.69314718055994531f;   // ln 2
constexpr float BETA2 = 0.28768207245178085f;   // ln(4/3)
constexpr float NEGS  = 0.2f;

typedef __attribute__((ext_vector_type(16))) __bf16 bf16x16;
typedef __attribute__((ext_vector_type(8)))  __bf16 bf16x8;
typedef __attribute__((ext_vector_type(4)))  __bf16 bf16x4;
typedef __attribute__((ext_vector_type(2)))  __bf16 bf16x2;
typedef __attribute__((ext_vector_type(8)))  float  f32x8;

// ---------------------------------------------------------------- helpers
__device__ __forceinline__ float leaky(float x) { return x > 0.f ? x : NEGS * x; }

__device__ __forceinline__ void atomicMaxF(float* addr, float val) {
    unsigned int* ua = (unsigned int*)addr;
    unsigned int old = *ua;
    while (__uint_as_float(old) < val) {
        unsigned int assumed = old;
        old = atomicCAS(ua, assumed, __float_as_uint(val));
        if (old == assumed) break;
    }
}

// ---------------------------------------------------------------- GEMM (bf16 WMMA)
// C[M,N] = act( A[M,K] @ B + bias ), A,B,C fp32 in memory, bf16 in the MACs.
// transB=0: B is [K,N] row-major.   transB=1: B is [N,K] row-major (i.e. @ W.T).
// M % 64 == 0, N % 64 == 0, K % 32 == 0 (true for every GEMM in this net).
//
// LDS layout: both tiles K-contiguous per row (B stored N-major) so every
// WMMA fragment is 2-4 ds_load_b128, matching the ISA 16-bit A/B layouts:
//   A lane(m,hi):  a[0..7]=K(hi*8..+7),  a[8..15]=K(16+hi*8..+7)
//   B lane(n,hi):  b[0..7]=K(hi*8..+7),  b[8..15]=K(16+hi*8..+7) at column n
constexpr int LSTR = 40;   // LDS row stride in bf16 elems (80B: 16B-aligned, bank-spread)

__global__ __launch_bounds__(256) void k_gemm_bf16(
    const float* __restrict__ A, const float* __restrict__ Bw,
    const float* __restrict__ bias, float* __restrict__ C,
    int M, int N, int K, int transB, int relu)
{
    __shared__ __bf16 As[64 * LSTR];   // 64 M-rows  x 32 K
    __shared__ __bf16 Bs[64 * LSTR];   // 64 N-cols  x 32 K   (N-major!)

    const int tid  = threadIdx.x;
    const int wave = tid >> 5;
    const int lane = tid & 31;
    const int bm   = blockIdx.x * 64;
    const int bn   = blockIdx.y * 64;
    const int wm   = (wave & 3) * 16;        // wave M tile
    const int wn   = (wave >> 2) * 32;       // wave N tile (two 16-wide subtiles)

    f32x8 acc0 = {};
    f32x8 acc1 = {};

    const int m   = lane & 15;
    const int hi  = lane >> 4;
    const int n   = lane & 15;
    const int kbb = hi * 8;

    for (int k0 = 0; k0 < K; k0 += 32) {
        // ---- stage A tile 64x32: float4 global loads -> packed bf16x4 LDS stores
        #pragma unroll
        for (int i = 0; i < 2; ++i) {
            int e = tid + i * 256;              // 0..511 float4s
            int r = e >> 3;                     // M row 0..63
            int c = (e & 7) * 4;                // K col 0,4,..,28
            float4 v = *(const float4*)&A[(size_t)(bm + r) * K + (k0 + c)];
            bf16x4 p;
            p[0] = (__bf16)v.x; p[1] = (__bf16)v.y;
            p[2] = (__bf16)v.z; p[3] = (__bf16)v.w;
            *(bf16x4*)&As[r * LSTR + c] = p;
        }
        // ---- stage B tile as Bs[n][k]
        if (transB) {
            // Bw[N,K]: rows already K-contiguous -> same vector path as A
            #pragma unroll
            for (int i = 0; i < 2; ++i) {
                int e = tid + i * 256;
                int r = e >> 3;                 // N col 0..63
                int c = (e & 7) * 4;            // K
                float4 v = *(const float4*)&Bw[(size_t)(bn + r) * K + (k0 + c)];
                bf16x4 p;
                p[0] = (__bf16)v.x; p[1] = (__bf16)v.y;
                p[2] = (__bf16)v.z; p[3] = (__bf16)v.w;
                *(bf16x4*)&Bs[r * LSTR + c] = p;
            }
        } else {
            // Bw[K,N]: transpose-stage, global loads coalesced along N,
            // packed K-pairs -> ds_store_b32
            #pragma unroll
            for (int i = 0; i < 4; ++i) {
                int e  = tid + i * 256;         // 0..1023 K-pairs
                int nc = e & 63;                // N col (fastest -> coalesced)
                int j  = e >> 6;                // K-pair 0..15
                float v0 = Bw[(size_t)(k0 + 2 * j)     * N + (bn + nc)];
                float v1 = Bw[(size_t)(k0 + 2 * j + 1) * N + (bn + nc)];
                bf16x2 p;
                p[0] = (__bf16)v0; p[1] = (__bf16)v1;
                *(bf16x2*)&Bs[nc * LSTR + 2 * j] = p;
            }
        }
        __syncthreads();

        // ---- fragments: pure b128 LDS loads
        bf16x8 alo = *(const bf16x8*)&As[(wm + m) * LSTR + hi * 8];
        bf16x8 ahi = *(const bf16x8*)&As[(wm + m) * LSTR + 16 + hi * 8];
        bf16x16 a = __builtin_shufflevector(alo, ahi,
                        0,1,2,3,4,5,6,7,8,9,10,11,12,13,14,15);

        bf16x8 b0lo = *(const bf16x8*)&Bs[(wn + n) * LSTR + kbb];
        bf16x8 b0hi = *(const bf16x8*)&Bs[(wn + n) * LSTR + kbb + 16];
        bf16x16 b0 = __builtin_shufflevector(b0lo, b0hi,
                        0,1,2,3,4,5,6,7,8,9,10,11,12,13,14,15);

        bf16x8 b1lo = *(const bf16x8*)&Bs[(wn + 16 + n) * LSTR + kbb];
        bf16x8 b1hi = *(const bf16x8*)&Bs[(wn + 16 + n) * LSTR + kbb + 16];
        bf16x16 b1 = __builtin_shufflevector(b1lo, b1hi,
                        0,1,2,3,4,5,6,7,8,9,10,11,12,13,14,15);

        acc0 = __builtin_amdgcn_wmma_f32_16x16x32_bf16(false, a, false, b0,
                                                       (short)0, acc0, false, false);
        acc1 = __builtin_amdgcn_wmma_f32_16x16x32_bf16(false, a, false, b1,
                                                       (short)0, acc1, false, false);
        __syncthreads();
    }

    // C/D layout: VGPR v -> M = v + 8*(lane/16); N = lane%16
    #pragma unroll
    for (int v = 0; v < 8; ++v) {
        int r  = bm + wm + v + 8 * hi;
        int c0 = bn + wn + n;
        int c1 = c0 + 16;
        float o0 = acc0[v], o1 = acc1[v];
        if (bias) { o0 += bias[c0]; o1 += bias[c1]; }
        if (relu) { o0 = fmaxf(o0, 0.f); o1 = fmaxf(o1, 0.f); }
        C[(size_t)r * N + c0] = o0;
        C[(size_t)r * N + c1] = o1;
    }
}

// ---------------------------------------------------------------- elementwise
__global__ void k_fill(float* p, float v, int n) {
    int i = blockIdx.x * blockDim.x + threadIdx.x;
    if (i < n) p[i] = v;
}
__global__ void k_copy(float* dst, const float* src, int n) {
    int i = blockIdx.x * blockDim.x + threadIdx.x;
    if (i < n) dst[i] = src[i];
}

// degree (self loop pre-counted via fill 1.0)
__global__ void k_deg(const int* __restrict__ col, float* deg, int e) {
    int i = blockIdx.x * blockDim.x + threadIdx.x;
    if (i < e) atomicAdd(&deg[col[i]], 1.f);
}
__global__ void k_dinv(const float* deg, float* dinv, int n) {
    int i = blockIdx.x * blockDim.x + threadIdx.x;
    if (i < n) dinv[i] = rsqrtf(deg[i]);
}
__global__ void k_norm(const int* row, const int* col, const float* dinv,
                       float* nrm, int e) {
    int i = blockIdx.x * blockDim.x + threadIdx.x;
    if (i < e) nrm[i] = dinv[row[i]] * dinv[col[i]];
}

// GCN edge aggregation: agg[col] += norm * h[row]   (one block per edge)
__global__ __launch_bounds__(256) void k_gcn_scatter(
    const float* __restrict__ h, const int* __restrict__ row,
    const int* __restrict__ col, const float* __restrict__ nrm,
    float* __restrict__ agg)
{
    int e = blockIdx.x;
    int r = row[e], c = col[e];
    float w = nrm[e];
    for (int d = threadIdx.x; d < DD; d += blockDim.x)
        atomicAdd(&agg[(size_t)c * DD + d], w * h[(size_t)r * DD + d]);
}
// T = (1-alpha)*(agg + dinv^2 * h) + alpha*h   (self loop folded in); in-place on agg
__global__ void k_gcn_T(float* agg, const float* h, const float* dinv, int n) {
    int i = blockIdx.x * blockDim.x + threadIdx.x;
    if (i >= n) return;
    int node = i / DD;
    float di = dinv[node];
    float t = agg[i] + di * di * h[i];
    agg[i] = (1.f - ALPHA) * t + ALPHA * h[i];
}
// h_out = relu((1-beta)*T + beta*U + res)
__global__ void k_gcn_fin(const float* T, const float* U, const float* res,
                          float* out, float beta, int n) {
    int i = blockIdx.x * blockDim.x + threadIdx.x;
    if (i < n) out[i] = fmaxf((1.f - beta) * T[i] + beta * U[i] + res[i], 0.f);
}

// per (node, head) attention scores: out[n,h] = sum_k xs[n,h*128+k]*att[h,k]
__global__ void k_headscore(const float* __restrict__ xs,
                            const float* __restrict__ att, float* out) {
    int idx = blockIdx.x * blockDim.x + threadIdx.x;
    if (idx >= NN * HH) return;
    int nidx = idx >> 2, h = idx & 3;
    const float4* xp = (const float4*)(xs + (size_t)nidx * DD + h * DGH);
    const float4* ap = (const float4*)(att + h * DGH);
    float acc = 0.f;
    #pragma unroll 4
    for (int k = 0; k < DGH / 4; ++k) {
        float4 a = xp[k], b = ap[k];
        acc += a.x * b.x + a.y * b.y + a.z * b.z + a.w * b.w;
    }
    out[idx] = acc;
}
// self-edge logit + running max init
__global__ void k_gat_self(const float* asrc, const float* adst,
                           float* selfE, float* mh) {
    int idx = blockIdx.x * blockDim.x + threadIdx.x;
    if (idx >= NN * HH) return;
    float v = leaky(asrc[idx] + adst[idx]);
    selfE[idx] = v; mh[idx] = v;
}
__global__ void k_gat_edge_max(const int* row, const int* col,
                               const float* asrc, const float* adst,
                               float* el, float* mh) {
    int idx = blockIdx.x * blockDim.x + threadIdx.x;
    if (idx >= EDG * HH) return;
    int e = idx >> 2, h = idx & 3;
    int r = row[e], c = col[e];
    float v = leaky(asrc[r * HH + h] + adst[c * HH + h]);
    el[idx] = v;
    atomicMaxF(&mh[c * HH + h], v);
}
__global__ void k_gat_sinit(const float* selfE, const float* mh, float* sh) {
    int idx = blockIdx.x * blockDim.x + threadIdx.x;
    if (idx < NN * HH) sh[idx] = expf(selfE[idx] - mh[idx]);
}
__global__ void k_gat_edge_exp(const int* col, const float* el, const float* mh,
                               float* ee, float* sh) {
    int idx = blockIdx.x * blockDim.x + threadIdx.x;
    if (idx >= EDG * HH) return;
    int e = idx >> 2, h = idx & 3;
    int c = col[e];
    float v = expf(el[idx] - mh[c * HH + h]);
    ee[idx] = v;
    atomicAdd(&sh[c * HH + h], v);
}
// h3 init with self-loop contribution
__global__ __launch_bounds__(256) void k_gat_h3_init(
    const float* selfE, const float* mh, const float* sh,
    const float* __restrict__ xs, float* __restrict__ h3)
{
    int nidx = blockIdx.x;
    for (int d = threadIdx.x; d < DD; d += blockDim.x) {
        int h = d >> 7;
        float w = expf(selfE[nidx * HH + h] - mh[nidx * HH + h]) / sh[nidx * HH + h];
        h3[(size_t)nidx * DD + d] = w * xs[(size_t)nidx * DD + d];
    }
}
__global__ __launch_bounds__(256) void k_gat_scatter(
    const int* row, const int* col, const float* ee, const float* sh,
    const float* __restrict__ xs, float* __restrict__ h3)
{
    int e = blockIdx.x;
    int r = row[e], c = col[e];
    float w[HH];
    #pragma unroll
    for (int h = 0; h < HH; ++h) w[h] = ee[e * HH + h] / sh[c * HH + h];
    for (int d = threadIdx.x; d < DD; d += blockDim.x)
        atomicAdd(&h3[(size_t)c * DD + d], w[d >> 7] * xs[(size_t)r * DD + d]);
}
__global__ void k_gat_fin(float* h3, const float* gat_b, int n) {
    int i = blockIdx.x * blockDim.x + threadIdx.x;
    if (i < n) h3[i] = fmaxf(h3[i] + gat_b[i % DD], 0.f);
}

// segment max of x[row] into xq[col] (xq pre-initialized to x = self loop)
__global__ __launch_bounds__(256) void k_xq_max(
    const int* row, const int* col, const float* __restrict__ x,
    float* __restrict__ xq)
{
    int e = blockIdx.x;
    int r = row[e], c = col[e];
    for (int d = threadIdx.x; d < DD; d += blockDim.x)
        atomicMaxF(&xq[(size_t)c * DD + d], x[(size_t)r * DD + d]);
}

// out[n] = dot(X[n,:512], w) (+ bias[0])
__global__ void k_dot512(const float* __restrict__ X, const float* __restrict__ w,
                         const float* bias, float* out, int n) {
    int i = blockIdx.x * blockDim.x + threadIdx.x;
    if (i >= n) return;
    const float4* xp = (const float4*)(X + (size_t)i * DD);
    const float4* wp = (const float4*)w;
    float acc = 0.f;
    #pragma unroll 4
    for (int k = 0; k < DD / 4; ++k) {
        float4 a = xp[k], b = wp[k];
        acc += a.x * b.x + a.y * b.y + a.z * b.z + a.w * b.w;
    }
    if (bias) acc += bias[0];
    out[i] = acc;
}

// ASAP softmax scalars (self edge folded into init)
__global__ void k_pool_self(const float* qa, const float* xa,
                            float* ssc, float* mp) {
    int i = blockIdx.x * blockDim.x + threadIdx.x;
    if (i >= NN) return;
    float v = leaky(qa[i] + xa[i]);
    ssc[i] = v; mp[i] = v;
}
__global__ void k_pool_edge_max(const int* row, const int* col,
                                const float* qa, const float* xa,
                                float* sc, float* mp) {
    int e = blockIdx.x * blockDim.x + threadIdx.x;
    if (e >= EDG) return;
    float v = leaky(qa[col[e]] + xa[row[e]]);
    sc[e] = v;
    atomicMaxF(&mp[col[e]], v);
}
__global__ void k_pool_sinit(const float* ssc, const float* mp, float* sp) {
    int i = blockIdx.x * blockDim.x + threadIdx.x;
    if (i < NN) sp[i] = expf(ssc[i] - mp[i]);
}
__global__ void k_pool_edge_exp(const int* col, const float* sc, const float* mp,
                                float* se, float* sp) {
    int e = blockIdx.x * blockDim.x + threadIdx.x;
    if (e >= EDG) return;
    int c = col[e];
    float v = expf(sc[e] - mp[c]);
    se[e] = v;
    atomicAdd(&sp[c], v);
}
__global__ __launch_bounds__(256) void k_xc_init(
    const float* ssc, const float* mp, const float* sp,
    const float* __restrict__ x, float* __restrict__ xc)
{
    int nidx = blockIdx.x;
    float w = expf(ssc[nidx] - mp[nidx]) / sp[nidx];
    for (int d = threadIdx.x; d < DD; d += blockDim.x)
        xc[(size_t)nidx * DD + d] = w * x[(size_t)nidx * DD + d];
}
__global__ __launch_bounds__(256) void k_xc_scatter(
    const int* row, const int* col, const float* se, const float* sp,
    const float* __restrict__ x, float* __restrict__ xc)
{
    int e = blockIdx.x;
    int r = row[e], c = col[e];
    float w = se[e] / sp[c];
    for (int d = threadIdx.x; d < DD; d += blockDim.x)
        atomicAdd(&xc[(size_t)c * DD + d], w * x[(size_t)r * DD + d]);
}

// LEConv fitness
__global__ void k_fit_init(const float* la, const float* lb, float* fits) {
    int i = blockIdx.x * blockDim.x + threadIdx.x;
    if (i < NN) fits[i] = la[i] - lb[i];           // self edge
}
__global__ void k_fit_edge(const int* row, const int* col,
                           const float* la, const float* lb, float* fits) {
    int e = blockIdx.x * blockDim.x + threadIdx.x;
    if (e >= EDG) return;
    atomicAdd(&fits[col[e]], la[row[e]] - lb[col[e]]);
}
__global__ void k_fit_fin(const float* fits, const float* l3, float* fit) {
    int i = blockIdx.x * blockDim.x + threadIdx.x;
    if (i < NN) fit[i] = 1.f / (1.f + expf(-(fits[i] + l3[i])));
}

// per-graph top-k selection by rank counting (order-invariant downstream)
__global__ __launch_bounds__(512) void k_topk(const float* __restrict__ fit,
                                              int* __restrict__ sel) {
    __shared__ float sf[NPG];
    int g = blockIdx.x, t = threadIdx.x;
    if (t < NPG) sf[t] = fit[g * NPG + t];
    __syncthreads();
    if (t >= NPG) return;
    float fi = sf[t];
    int rank = 0;
    for (int j = 0; j < NPG; ++j)
        rank += (sf[j] > fi) || (sf[j] == fi && j < t);
    sel[g * NPG + t] = rank < KPOOL;
}

// mean pools -> feats[b, slot, :]
__global__ __launch_bounds__(256) void k_mean_all(const float* __restrict__ src,
                                                  float* feats, int slot) {
    int g = blockIdx.x;
    int d = blockIdx.y * 256 + threadIdx.x;
    float acc = 0.f;
    for (int i = 0; i < NPG; ++i)
        acc += src[(size_t)(g * NPG + i) * DD + d];
    feats[(size_t)(g * 3 + slot) * DD + d] = acc * (1.f / NPG);
}
__global__ __launch_bounds__(256) void k_mean_sel(const float* __restrict__ xc,
                                                  const float* __restrict__ fit,
                                                  const int* __restrict__ sel,
                                                  float* feats, int slot) {
    int g = blockIdx.x;
    int d = blockIdx.y * 256 + threadIdx.x;
    float acc = 0.f;
    for (int i = 0; i < NPG; ++i) {
        int nidx = g * NPG + i;
        if (sel[nidx])
            acc += fmaxf(xc[(size_t)nidx * DD + d] * fit[nidx], 0.f);
    }
    feats[(size_t)(g * 3 + slot) * DD + d] = acc * (1.f / KPOOL);
}

// 2-head attention over 3 tokens, head dim 256. blockIdx encodes (b,h,q).
__global__ __launch_bounds__(256) void k_attn(const float* __restrict__ qkv,
                                              float* __restrict__ o) {
    int b = blockIdx.x / 6, rem = blockIdx.x % 6;
    int h = rem / 3, q = rem % 3;
    int t = threadIdx.x;
    __shared__ float red[256];
    __shared__ float lg[3];
    const float* qv = qkv + (size_t)(b * 3 + q) * (3 * DD) + h * 256;
    for (int k = 0; k < 3; ++k) {
        const float* kv = qkv + (size_t)(b * 3 + k) * (3 * DD) + DD + h * 256;
        red[t] = qv[t] * kv[t];
        __syncthreads();
        for (int s = 128; s > 0; s >>= 1) {
            if (t < s) red[t] += red[t + s];
            __syncthreads();
        }
        if (t == 0) lg[k] = red[0] * (1.f / 16.f);   // 1/sqrt(256)
        __syncthreads();
    }
    float m = fmaxf(lg[0], fmaxf(lg[1], lg[2]));
    float p0 = expf(lg[0] - m), p1 = expf(lg[1] - m), p2 = expf(lg[2] - m);
    float inv = 1.f / (p0 + p1 + p2);
    const float* v0 = qkv + (size_t)(b * 3 + 0) * (3 * DD) + 2 * DD + h * 256;
    const float* v1 = qkv + (size_t)(b * 3 + 1) * (3 * DD) + 2 * DD + h * 256;
    const float* v2 = qkv + (size_t)(b * 3 + 2) * (3 * DD) + 2 * DD + h * 256;
    o[(size_t)(b * 3 + q) * DD + h * 256 + t] =
        (p0 * v0[t] + p1 * v1[t] + p2 * v2[t]) * inv;
}

__global__ void k_fused_mean(const float* __restrict__ aout, float* fused) {
    int i = blockIdx.x * blockDim.x + threadIdx.x;
    if (i >= BG * DD) return;
    int b = i / DD, d = i % DD;
    fused[i] = (aout[(size_t)(b * 3 + 0) * DD + d] +
                aout[(size_t)(b * 3 + 1) * DD + d] +
                aout[(size_t)(b * 3 + 2) * DD + d]) * (1.f / 3.f);
}

// ---------------------------------------------------------------- launch
extern "C" void kernel_launch(void* const* d_in, const int* in_sizes, int n_in,
                              void* d_out, int out_size, void* d_ws, size_t ws_size,
                              hipStream_t stream)
{
    (void)in_sizes; (void)n_in; (void)out_size; (void)ws_size;

    const float* X_IN  = (const float*)d_in[0];
    const int*   EIDX  = (const int*)  d_in[1];
    const float* LIN_W = (const float*)d_in[3];
    const float* LIN_B = (const float*)d_in[4];
    const float* GCN1W = (const float*)d_in[5];
    const float* GCN2W = (const float*)d_in[6];
    const float* GATW  = (const float*)d_in[7];
    const float* ASRCW = (const float*)d_in[8];
    const float* ADSTW = (const float*)d_in[9];
    const float* GATB  = (const float*)d_in[10];
    const float* PLW   = (const float*)d_in[11];
    const float* PLB   = (const float*)d_in[12];
    const float* PAW   = (const float*)d_in[13];
    const float* PAB   = (const float*)d_in[14];
    const float* LE1W  = (const float*)d_in[15];
    const float* LE1B  = (const float*)d_in[16];
    const float* LE2W  = (const float*)d_in[17];
    const float* LE3W  = (const float*)d_in[18];
    const float* LE3B  = (const float*)d_in[19];
    const float* MIW   = (const float*)d_in[20];
    const float* MIB   = (const float*)d_in[21];
    const float* MOW   = (const float*)d_in[22];
    const float* MOB   = (const float*)d_in[23];
    const float* CLSW  = (const float*)d_in[24];
    const float* CLSB  = (const float*)d_in[25];

    const int* ROW = EIDX;          // edge sources j
    const int* COL = EIDX + EDG;    // edge targets i

    // ---- workspace carve
    size_t cur = 0;
    auto carve = [&](size_t nfloats) -> float* {
        float* p = (float*)((char*)d_ws + cur);
        cur += (nfloats * sizeof(float) + 255) & ~(size_t)255;
        return p;
    };
    float* X    = carve((size_t)NN * DD);   // relu(lin(x))
    float* T    = carve((size_t)NN * DD);   // gcn agg/T ; later xq
    float* U    = carve((size_t)NN * DD);   // gemm out  ; later h3
    float* W1   = carve((size_t)NN * DD);   // h1 ; later xs ; later q
    float* H2   = carve((size_t)NN * DD);
    float* XC   = carve((size_t)NN * DD);
    float* DEG  = carve(NN);
    float* DINV = carve(NN);
    float* NRM  = carve(EDG);
    float* ASRC = carve((size_t)NN * HH);
    float* ADST = carve((size_t)NN * HH);
    float* SELFE= carve((size_t)NN * HH);
    float* MH   = carve((size_t)NN * HH);
    float* SH   = carve((size_t)NN * HH);
    float* EL   = carve((size_t)EDG * HH);
    float* EEB  = carve((size_t)EDG * HH);
    float* QA   = carve(NN);
    float* XA   = carve(NN);
    float* SSC  = carve(NN);
    float* MP   = carve(NN);
    float* SP   = carve(NN);
    float* SCE  = carve(EDG);
    float* SEE  = carve(EDG);
    float* LA   = carve(NN);
    float* LB   = carve(NN);
    float* L3   = carve(NN);
    float* FITS = carve(NN);
    float* FIT  = carve(NN);
    int*   SEL  = (int*)carve(NN);
    float* FEATS= carve((size_t)BG * 3 * DD);
    float* QKV  = carve((size_t)BG * 3 * 3 * DD);
    float* ATO  = carve((size_t)BG * 3 * DD);
    float* AOUT = carve((size_t)BG * 3 * DD);

    auto gemm = [&](const float* A, const float* Bw, const float* bias, float* C,
                    int M, int N, int K, int transB, int relu) {
        dim3 g(M / 64, N / 64);
        k_gemm_bf16<<<g, 256, 0, stream>>>(A, Bw, bias, C, M, N, K, transB, relu);
    };
    const int ND   = NN * DD;
    const dim3 bND((ND + 255) / 256);
    const dim3 bNH((NN * HH + 255) / 256);
    const dim3 bEH((EDG * HH + 255) / 256);
    const dim3 bE ((EDG + 255) / 256);
    const dim3 bN ((NN + 255) / 256);

    // 1) x = relu(x_in @ lin_w + lin_b)   [19200,1024]x[1024,512] (WMMA)
    gemm(X_IN, LIN_W, LIN_B, X, NN, DD, DIN, 0, 1);

    // 2) degrees / symmetric norm (self loop = +1 via init)
    k_fill<<<bN, 256, 0, stream>>>(DEG, 1.f, NN);
    k_deg<<<bE, 256, 0, stream>>>(COL, DEG, EDG);
    k_dinv<<<bN, 256, 0, stream>>>(DEG, DINV, NN);
    k_norm<<<bE, 256, 0, stream>>>(ROW, COL, DINV, NRM, EDG);

    // 3) GCN2Conv layer 1:  h1 = relu(gcn2(x,x) + x)
    k_fill<<<bND, 256, 0, stream>>>(T, 0.f, ND);
    k_gcn_scatter<<<EDG, 256, 0, stream>>>(X, ROW, COL, NRM, T);
    k_gcn_T<<<bND, 256, 0, stream>>>(T, X, DINV, ND);
    gemm(T, GCN1W, nullptr, U, NN, DD, DD, 0, 0);                 // WMMA
    k_gcn_fin<<<bND, 256, 0, stream>>>(T, U, X, W1, BETA1, ND);   // W1 = h1

    // 4) GCN2Conv layer 2:  h2 = relu(gcn2(h1,h1) + h1)
    k_fill<<<bND, 256, 0, stream>>>(T, 0.f, ND);
    k_gcn_scatter<<<EDG, 256, 0, stream>>>(W1, ROW, COL, NRM, T);
    k_gcn_T<<<bND, 256, 0, stream>>>(T, W1, DINV, ND);
    gemm(T, GCN2W, nullptr, U, NN, DD, DD, 0, 0);                 // WMMA
    k_gcn_fin<<<bND, 256, 0, stream>>>(T, U, W1, H2, BETA2, ND);

    // 5) GAT: xs = x @ gat_w (reuse W1), segment softmax over col, h3 (reuse U)
    gemm(X, GATW, nullptr, W1, NN, DD, DD, 0, 0);                 // WMMA
    k_headscore<<<bNH, 256, 0, stream>>>(W1, ASRCW, ASRC);
    k_headscore<<<bNH, 256, 0, stream>>>(W1, ADSTW, ADST);
    k_gat_self<<<bNH, 256, 0, stream>>>(ASRC, ADST, SELFE, MH);
    k_gat_edge_max<<<bEH, 256, 0, stream>>>(ROW, COL, ASRC, ADST, EL, MH);
    k_gat_sinit<<<bNH, 256, 0, stream>>>(SELFE, MH, SH);
    k_gat_edge_exp<<<bEH, 256, 0, stream>>>(COL, EL, MH, EEB, SH);
    k_gat_h3_init<<<NN, 256, 0, stream>>>(SELFE, MH, SH, W1, U);
    k_gat_scatter<<<EDG, 256, 0, stream>>>(ROW, COL, EEB, SH, W1, U);
    k_gat_fin<<<bND, 256, 0, stream>>>(U, GATB, ND);              // U = h3

    // 6) ASAP pooling: xq = segment_max(x[row]) with self loop = x
    k_copy<<<bND, 256, 0, stream>>>(T, X, ND);                    // T = xq
    k_xq_max<<<EDG, 256, 0, stream>>>(ROW, COL, X, T);
    gemm(T, PLW, PLB, W1, NN, DD, DD, 0, 0);                      // W1 = q (WMMA)
    k_dot512<<<bN, 256, 0, stream>>>(W1, PAW, PAB, QA, NN);       // q . w0 + b
    k_dot512<<<bN, 256, 0, stream>>>(X, PAW + DD, nullptr, XA, NN);
    k_pool_self<<<bN, 256, 0, stream>>>(QA, XA, SSC, MP);
    k_pool_edge_max<<<bE, 256, 0, stream>>>(ROW, COL, QA, XA, SCE, MP);
    k_pool_sinit<<<bN, 256, 0, stream>>>(SSC, MP, SP);
    k_pool_edge_exp<<<bE, 256, 0, stream>>>(COL, SCE, MP, SEE, SP);
    k_xc_init<<<NN, 256, 0, stream>>>(SSC, MP, SP, X, XC);
    k_xc_scatter<<<EDG, 256, 0, stream>>>(ROW, COL, SEE, SP, X, XC);

    // 7) LEConv fitness + sigmoid
    k_dot512<<<bN, 256, 0, stream>>>(XC, LE1W, LE1B, LA, NN);
    k_dot512<<<bN, 256, 0, stream>>>(XC, LE2W, nullptr, LB, NN);
    k_dot512<<<bN, 256, 0, stream>>>(XC, LE3W, LE3B, L3, NN);
    k_fit_init<<<bN, 256, 0, stream>>>(LA, LB, FITS);
    k_fit_edge<<<bE, 256, 0, stream>>>(ROW, COL, LA, LB, FITS);
    k_fit_fin<<<bN, 256, 0, stream>>>(FITS, L3, FIT);

    // 8) per-graph top-k selection + pooled views
    k_topk<<<BG, 512, 0, stream>>>(FIT, SEL);
    k_mean_all<<<dim3(BG, DD / 256), 256, 0, stream>>>(H2, FEATS, 0);
    k_mean_all<<<dim3(BG, DD / 256), 256, 0, stream>>>(U, FEATS, 1);    // h3
    k_mean_sel<<<dim3(BG, DD / 256), 256, 0, stream>>>(XC, FIT, SEL, FEATS, 2);

    // 9) MHA over 3 pooled views (qkv/out projections via WMMA)
    gemm(FEATS, MIW, MIB, QKV, BG * 3, 3 * DD, DD, 1, 0);         // @ W.T
    k_attn<<<BG * 2 * 3, 256, 0, stream>>>(QKV, ATO);
    gemm(ATO, MOW, MOB, AOUT, BG * 3, DD, DD, 1, 0);              // @ W.T
    k_fused_mean<<<(BG * DD + 255) / 256, 256, 0, stream>>>(AOUT, (float*)d_out);

    // 10) classifier: logit = fused @ cls_w + cls_b   -> d_out[64*512 ...]
    gemm((float*)d_out, CLSW, CLSB, (float*)d_out + BG * DD, BG, NC, DD, 0, 0);
}
